// simplemodel_14611478741208
// MI455X (gfx1250) — compile-verified, matrix operand order checked by reading, not run on hardware
//
#include <hip/hip_runtime.h>
#include <hip/hip_fp16.h>
#include <hip/hip_bf16.h>

// ---------------------------------------------------------------------------
// Model dims (fixed by the reference)
// ---------------------------------------------------------------------------
#define B_  16
#define S_  512
#define D_  1024
#define H_  128      // lstm hidden; 4H = 512, 2H = 256
#define G_  256      // gat dim
#define HP_ 136      // padded h row stride (halfs) to break LDS bank conflicts
#define ZP_ 516      // padded z row stride (floats)
// Workspace requirement: ~101 MB, assumed available.

typedef __attribute__((ext_vector_type(16))) _Float16 v16h;
typedef __attribute__((ext_vector_type(8)))  float    v8f;

// ---------------------------------------------------------------------------
// Fragment loaders.
// A (16x32, f16): lane L<16 holds row M=L, K = k0+{0..7, 16..23};
//                 lane L>=16 holds row M=L-16, K = k0+{8..15, 24..31}.
// B (32x16, f16): column-major == rows of W; same per-lane K packing.
// ---------------------------------------------------------------------------
__device__ __forceinline__ v16h frag_from_f32(const float* p) {
  // p already offset to k0 + (lane&16 ? 8 : 0); reads p[0..7] and p[16..23]
  float4 x0 = *(const float4*)(p);
  float4 x1 = *(const float4*)(p + 4);
  float4 y0 = *(const float4*)(p + 16);
  float4 y1 = *(const float4*)(p + 20);
  v16h f;
  f[0]=(_Float16)x0.x;  f[1]=(_Float16)x0.y;  f[2]=(_Float16)x0.z;  f[3]=(_Float16)x0.w;
  f[4]=(_Float16)x1.x;  f[5]=(_Float16)x1.y;  f[6]=(_Float16)x1.z;  f[7]=(_Float16)x1.w;
  f[8]=(_Float16)y0.x;  f[9]=(_Float16)y0.y;  f[10]=(_Float16)y0.z; f[11]=(_Float16)y0.w;
  f[12]=(_Float16)y1.x; f[13]=(_Float16)y1.y; f[14]=(_Float16)y1.z; f[15]=(_Float16)y1.w;
  return f;
}

__device__ __forceinline__ v16h frag_from_f16(const _Float16* p) {
  v16h f;
  #pragma unroll
  for (int i = 0; i < 8; ++i) { f[i] = p[i]; f[8 + i] = p[16 + i]; }
  return f;
}

#define WMMA_F16(a, b, c) __builtin_amdgcn_wmma_f32_16x16x32_f16( \
    false, (a), false, (b), (short)0, (c), false, false)

// ---------------------------------------------------------------------------
// WMMA GEMM:  C = act(A @ B^T + bias + bias2) [+ resid]
//   B is [N, K] row-major (weight-style); all call sites use this layout.
//   Register blocking: each wave computes a 32x64 output patch
//   (2 M-tiles x 4 N-tiles): A fragments reused 4x, B fragments 2x.
//   Template params (compile-time -> branch-free inner loop):
//     A_MODE: 0 plain rows; 1 = LSTM xs layout (row m -> x[(m&15)*S + (m>>4)])
//     ACT:    0 none, 1 relu, 2 leaky(0.1)
//     FULLN:  N % 64 == 0 (no column guards/clamps anywhere)
//   grid: (ceil(N/256), ceil(M/32), batch); block: 128 (4 waves)
// ---------------------------------------------------------------------------
template <int A_MODE, int ACT, bool FULLN>
__global__ __launch_bounds__(128) void gemm_wmma_kernel(
    const float* __restrict__ A, const float* __restrict__ Bm,
    const float* __restrict__ bias, const float* __restrict__ bias2,
    const float* __restrict__ resid, float* __restrict__ C,
    int M, int N, int K, int lda, int ldb, int ldc,
    long long sA, long long sB, long long sC)
{
  const int lane  = threadIdx.x & 31;
  const int wave  = threadIdx.x >> 5;
  const int tileM = blockIdx.y * 32;
  const int nbase = blockIdx.x * 256 + wave * 64;
  if (nbase >= N) return;                       // wave-uniform exit

  const int z = blockIdx.z;
  const float* Ab = A  + (size_t)z * (size_t)sA;
  const float* Bb = Bm + (size_t)z * (size_t)sB;
  float*       Cb = C  + (size_t)z * (size_t)sC;
  const float* Rb = resid ? (resid + (size_t)z * (size_t)sC) : nullptr;

  const int l15   = lane & 15;
  const int khalf = (lane & 16) ? 8 : 0;

  int am0 = tileM + l15;      if (am0 >= M) am0 = M - 1;     // clamp loads
  int am1 = tileM + 16 + l15; if (am1 >= M) am1 = M - 1;
  size_t arow0, arow1;
  if (A_MODE == 1) {
    arow0 = ((size_t)(am0 & 15) * S_ + (size_t)(am0 >> 4)) * (size_t)lda;
    arow1 = ((size_t)(am1 & 15) * S_ + (size_t)(am1 >> 4)) * (size_t)lda;
  } else {
    arow0 = (size_t)am0 * (size_t)lda;
    arow1 = (size_t)am1 * (size_t)lda;
  }

  bool ton[4]; int bn[4];
  #pragma unroll
  for (int nt = 0; nt < 4; ++nt) {
    if (FULLN) {
      ton[nt] = true;
      bn[nt]  = nbase + nt * 16 + l15;
    } else {
      ton[nt] = (nbase + nt * 16) < N;          // wave-uniform
      int c = nbase + nt * 16 + l15; if (c >= N) c = N - 1;
      bn[nt] = c;
    }
  }

  v8f acc[2][4] = {};
  for (int k0 = 0; k0 < K; k0 += 32) {
    const float* pa0 = Ab + arow0 + k0 + khalf;
    const float* pa1 = Ab + arow1 + k0 + khalf;
    v16h a0 = frag_from_f32(pa0);
    v16h a1 = frag_from_f32(pa1);
    if (k0 + 32 < K) {
      __builtin_prefetch(pa0 + 32, 0, 1);
      __builtin_prefetch(pa1 + 32, 0, 1);
    }
    #pragma unroll
    for (int nt = 0; nt < 4; ++nt) {
      if (!FULLN && !ton[nt]) continue;
      v16h b = frag_from_f32(Bb + (size_t)bn[nt] * (size_t)ldb + k0 + khalf);
      acc[0][nt] = WMMA_F16(a0, b, acc[0][nt]);
      acc[1][nt] = WMMA_F16(a1, b, acc[1][nt]);
    }
  }

  const int rb = (lane < 16) ? 0 : 8;
  #pragma unroll
  for (int nt = 0; nt < 4; ++nt) {
    if (!FULLN && !ton[nt]) continue;
    const int col = nbase + nt * 16 + l15;
    bool colok = FULLN ? true : (col < N);
    float bv = 0.0f;
    if (colok) {
      if (bias)  bv += bias[col];
      if (bias2) bv += bias2[col];
    }
    #pragma unroll
    for (int mt = 0; mt < 2; ++mt) {
      #pragma unroll
      for (int r = 0; r < 8; ++r) {
        const int row = tileM + mt * 16 + rb + r;
        if (row < M && colok) {
          float v = acc[mt][nt][r] + bv;
          if      (ACT == 1) v = v > 0.0f ? v : 0.0f;
          else if (ACT == 2) v = v > 0.0f ? v : 0.1f * v;
          if (Rb) v += Rb[(size_t)row * (size_t)ldc + col];
          Cb[(size_t)row * (size_t)ldc + col] = v;
        }
      }
    }
  }
}

// ---------------------------------------------------------------------------
// f32 -> f16 weight conversion (Whh copies for the recurrence)
// ---------------------------------------------------------------------------
__global__ void cvt_f32_f16_kernel(const float* __restrict__ src,
                                   __half* __restrict__ dst, int n) {
  int i = blockIdx.x * blockDim.x + threadIdx.x;
  if (i < n) ((_Float16*)dst)[i] = (_Float16)src[i];
}

// ---------------------------------------------------------------------------
// Batched transpose: src [B, S, G] -> dst [B, G, S] (for a@g aggregation,
// converts strided B-fragment loads into contiguous b128 loads).
// grid (S/32, G/32, B), block (32, 8)
// ---------------------------------------------------------------------------
__global__ __launch_bounds__(256) void transpose_bsg_kernel(
    const float* __restrict__ src, float* __restrict__ dst)
{
  __shared__ float tile[32][33];
  const int b  = blockIdx.z;
  const int s0 = blockIdx.x * 32, g0 = blockIdx.y * 32;
  const int tx = threadIdx.x, ty = threadIdx.y;
  const float* sp = src + (size_t)b * S_ * G_;
  float*       dp = dst + (size_t)b * G_ * S_;
  #pragma unroll
  for (int r = 0; r < 32; r += 8)
    tile[ty + r][tx] = sp[(size_t)(s0 + ty + r) * G_ + g0 + tx];
  __syncthreads();
  #pragma unroll
  for (int r = 0; r < 32; r += 8)
    dp[(size_t)(g0 + ty + r) * S_ + s0 + tx] = tile[tx][ty + r];
}

// ---------------------------------------------------------------------------
// LSTM recurrence, one persistent workgroup per direction (blockIdx.x).
// pre = x@Wih^T + bih + bhh, laid out [S, B, 4H].  hs out: [S, B, 2H].
// Whh fragments (16 per wave) stay resident in VGPRs for all 512 steps;
// per step: 4 LDS A-fragment reads feed 16 WMMAs per wave.
// ---------------------------------------------------------------------------
__global__ __launch_bounds__(256) void lstm_dir_kernel(
    const float* __restrict__ preF, const float* __restrict__ preB,
    const __half* __restrict__ whhF16_f, const __half* __restrict__ whhF16_b,
    float* __restrict__ hs)
{
  const int dir = blockIdx.x;
  const float*    pre = dir ? preB : preF;
  const _Float16* whh = (const _Float16*)(dir ? whhF16_b : whhF16_f);

  __shared__ _Float16 h_lds[B_ * HP_];       // padded: ~4.3 KB
  __shared__ float    c_lds[B_ * H_];        // 8 KB
  __shared__ float    z_lds[B_ * ZP_];       // padded: ~33 KB

  const int tid   = threadIdx.x;
  const int wave  = tid >> 5;
  const int lane  = tid & 31;
  const int l15   = lane & 15;
  const int khalf = (lane & 16) ? 8 : 0;

  // resident Whh fragments: [nt][kc], 128 VGPRs/lane
  v16h wfrag[16];
  #pragma unroll
  for (int nt = 0; nt < 4; ++nt) {
    const int tileN = (wave * 4 + nt) * 16;
    #pragma unroll
    for (int kc = 0; kc < 4; ++kc)
      wfrag[nt * 4 + kc] =
          frag_from_f16(whh + (size_t)(tileN + l15) * H_ + kc * 32 + khalf);
  }

  for (int i = tid; i < B_ * HP_; i += 256) h_lds[i] = (_Float16)0.0f;
  for (int i = tid; i < B_ * H_;  i += 256) c_lds[i] = 0.0f;
  __syncthreads();

  for (int step = 0; step < S_; ++step) {
    const int t = dir ? (S_ - 1 - step) : step;
    const float* pre_t = pre + (size_t)t * (B_ * 4 * H_);

    // A fragments from h (shared across all 4 N-tiles)
    v16h a[4];
    #pragma unroll
    for (int kc = 0; kc < 4; ++kc)
      a[kc] = frag_from_f16(&h_lds[l15 * HP_ + kc * 32 + khalf]);

    #pragma unroll
    for (int nt = 0; nt < 4; ++nt) {
      v8f acc = {};
      #pragma unroll
      for (int kc = 0; kc < 4; ++kc)
        acc = WMMA_F16(a[kc], wfrag[nt * 4 + kc], acc);
      const int col = (wave * 4 + nt) * 16 + l15;
      const int rb  = (lane < 16) ? 0 : 8;
      #pragma unroll
      for (int r = 0; r < 8; ++r) {
        const int row = rb + r;                       // batch index
        z_lds[row * ZP_ + col] = acc[r] + pre_t[row * 512 + col];
      }
    }
    __syncthreads();

    // gates: z = [i | f | g | o] along the 4H axis
    for (int i = tid; i < B_ * H_; i += 256) {
      const int b_ = i >> 7, hx = i & (H_ - 1);
      const float zi = z_lds[b_ * ZP_ + hx];
      const float zf = z_lds[b_ * ZP_ + 128 + hx];
      const float zg = z_lds[b_ * ZP_ + 256 + hx];
      const float zo = z_lds[b_ * ZP_ + 384 + hx];
      const float si = 1.0f / (1.0f + __expf(-zi));
      const float sf = 1.0f / (1.0f + __expf(-zf));
      const float so = 1.0f / (1.0f + __expf(-zo));
      float c = sf * c_lds[i] + si * tanhf(zg);
      const float h = so * tanhf(c);
      c_lds[i] = c;
      h_lds[b_ * HP_ + hx] = (_Float16)h;
      hs[((size_t)t * B_ + b_) * (2 * H_) + dir * H_ + hx] = h;
    }
    __syncthreads();
  }
}

// ---------------------------------------------------------------------------
// Attention post-processing: a = (sigmoid(s) + 1e-5*I) * adj(diag=1), then
// row-normalize in place.  grid: (S, B), block: 256.
// ---------------------------------------------------------------------------
__global__ __launch_bounds__(256) void attn_mask_norm_kernel(
    float* __restrict__ scores, const float* __restrict__ adj)
{
  const int i = blockIdx.x, b = blockIdx.y, tid = threadIdx.x;
  float* row        = scores + ((size_t)b * S_ + i) * S_;
  const float* arow = adj    + ((size_t)b * S_ + i) * S_;
  __shared__ float red[256];
  float part = 0.0f;
  for (int j = tid; j < S_; j += 256) {
    const float sg = 1.0f / (1.0f + __expf(-row[j]));
    const float av = (j == i) ? 1.0f : arow[j];
    const float v  = (sg + ((j == i) ? 1e-5f : 0.0f)) * av;
    row[j] = v;
    part += v;
  }
  red[tid] = part;
  __syncthreads();
  for (int off = 128; off > 0; off >>= 1) {
    if (tid < off) red[tid] += red[tid + off];
    __syncthreads();
  }
  const float inv = 1.0f / red[0];
  for (int j = tid; j < S_; j += 256) row[j] *= inv;
}

// ---------------------------------------------------------------------------
// hidden[b, 0:256]  = mean_s h1[s, b, :]   (h1 layout [S,B,2H])
// hidden[b, 256:512]= mean_s gfin[b, s, :] (gfin layout [B,S,G])
// grid: 16, block: 256
// ---------------------------------------------------------------------------
__global__ __launch_bounds__(256) void head_mean_kernel(
    const float* __restrict__ h1, const float* __restrict__ gfin,
    float* __restrict__ hidden)
{
  const int b = blockIdx.x, tid = threadIdx.x;
  for (int d = tid; d < 512; d += 256) {
    float s = 0.0f;
    if (d < 256) {
      for (int t = 0; t < S_; ++t) s += h1[((size_t)t * B_ + b) * 256 + d];
    } else {
      const int dd = d - 256;
      for (int t = 0; t < S_; ++t) s += gfin[((size_t)b * S_ + t) * 256 + dd];
    }
    hidden[b * 512 + d] = s * (1.0f / (float)S_);
  }
}

// ---------------------------------------------------------------------------
// Host-side launcher
// ---------------------------------------------------------------------------
template <int A_MODE, int ACT, bool FULLN>
static inline void gemm(hipStream_t st,
                        const float* A, const float* Bm,
                        const float* bias, const float* bias2, const float* resid,
                        float* C, int M, int N, int K, int lda, int ldb, int ldc,
                        long long sA = 0, long long sB = 0, long long sC = 0,
                        int batches = 1)
{
  dim3 grid((N + 255) / 256, (M + 31) / 32, batches);
  gemm_wmma_kernel<A_MODE, ACT, FULLN><<<grid, 128, 0, st>>>(
      A, Bm, bias, bias2, resid, C, M, N, K, lda, ldb, ldc, sA, sB, sC);
}

extern "C" void kernel_launch(void* const* d_in, const int* in_sizes, int n_in,
                              void* d_out, int out_size, void* d_ws, size_t ws_size,
                              hipStream_t stream)
{
  (void)in_sizes; (void)n_in; (void)out_size; (void)ws_size;
  auto F = [&](int i) { return (const float*)d_in[i]; };

  const float* x        = F(0);
  const float* adj      = F(1);
  const float* l0f_Wih  = F(2),  *l0f_Whh = F(3),  *l0f_bih = F(4),  *l0f_bhh = F(5);
  const float* l0b_Wih  = F(6),  *l0b_Whh = F(7),  *l0b_bih = F(8),  *l0b_bhh = F(9);
  const float* l1f_Wih  = F(10), *l1f_Whh = F(11), *l1f_bih = F(12), *l1f_bhh = F(13);
  const float* l1b_Wih  = F(14), *l1b_Whh = F(15), *l1b_bih = F(16), *l1b_bhh = F(17);
  const float* fc1_W    = F(18), *fc1_b   = F(19);
  const float* wattn_W  = F(20), *wattn_b = F(21);
  const float* lin0_W   = F(22), *lin0_b  = F(23);
  const float* lin1_W   = F(24), *lin1_b  = F(25);
  const float* linf_W   = F(26), *linf_b  = F(27);
  const float* fcA_W    = F(28), *fcA_b   = F(29);
  const float* fcB_W    = F(30), *fcB_b   = F(31);
  const float* fcC_W    = F(32), *fcC_b   = F(33);

  // ---- workspace carve-out (256B aligned slices) ----
  char* wsp = (char*)d_ws;
  size_t off = 0;
  auto carve = [&](size_t bytes) -> void* {
    void* p = wsp + off;
    off += (bytes + 255) & ~(size_t)255;
    return p;
  };
  float* preF   = (float*)carve((size_t)S_ * B_ * 4 * H_ * 4);  // 16 MB
  float* preB   = (float*)carve((size_t)S_ * B_ * 4 * H_ * 4);  // 16 MB
  float* h0     = (float*)carve((size_t)S_ * B_ * 2 * H_ * 4);  //  8 MB
  float* h1     = (float*)carve((size_t)S_ * B_ * 2 * H_ * 4);  //  8 MB
  float* gA     = (float*)carve((size_t)B_ * S_ * G_ * 4);      //  8 MB
  float* gB     = (float*)carve((size_t)B_ * S_ * G_ * 4);      //  8 MB
  float* gC     = (float*)carve((size_t)B_ * S_ * G_ * 4);      //  8 MB
  float* gT     = (float*)carve((size_t)B_ * G_ * S_ * 4);      //  8 MB
  float* scores = (float*)carve((size_t)B_ * S_ * S_ * 4);      // 16 MB
  __half* whh0f = (__half*)carve((size_t)4 * H_ * H_ * 2);
  __half* whh0b = (__half*)carve((size_t)4 * H_ * H_ * 2);
  __half* whh1f = (__half*)carve((size_t)4 * H_ * H_ * 2);
  __half* whh1b = (__half*)carve((size_t)4 * H_ * H_ * 2);
  float* fcAo   = (float*)carve((size_t)B_ * 128 * 4);
  float* fcBo   = (float*)carve((size_t)B_ * 32 * 4);

  const int MSB = S_ * B_;          // 8192
  const int nwh = 4 * H_ * H_;      // 65536

  // ---- Whh f16 copies (one-time per call, cheap) ----
  cvt_f32_f16_kernel<<<(nwh + 255) / 256, 256, 0, stream>>>(l0f_Whh, whh0f, nwh);
  cvt_f32_f16_kernel<<<(nwh + 255) / 256, 256, 0, stream>>>(l0b_Whh, whh0b, nwh);
  cvt_f32_f16_kernel<<<(nwh + 255) / 256, 256, 0, stream>>>(l1f_Whh, whh1f, nwh);
  cvt_f32_f16_kernel<<<(nwh + 255) / 256, 256, 0, stream>>>(l1b_Whh, whh1b, nwh);

  // ---- LSTM layer 0: pre = x@Wih^T + bih + bhh (rows ordered m = s*B + b) ----
  gemm<1, 0, true>(stream, x, l0f_Wih, l0f_bih, l0f_bhh, nullptr, preF,
                   MSB, 4 * H_, D_, D_, D_, 4 * H_);
  gemm<1, 0, true>(stream, x, l0b_Wih, l0b_bih, l0b_bhh, nullptr, preB,
                   MSB, 4 * H_, D_, D_, D_, 4 * H_);
  lstm_dir_kernel<<<2, 256, 0, stream>>>(preF, preB, whh0f, whh0b, h0);

  // ---- LSTM layer 1 (input h0 is [S,B,2H], already m = s*B + b contiguous) ----
  gemm<0, 0, true>(stream, h0, l1f_Wih, l1f_bih, l1f_bhh, nullptr, preF,
                   MSB, 4 * H_, 2 * H_, 2 * H_, 2 * H_, 4 * H_);
  gemm<0, 0, true>(stream, h0, l1b_Wih, l1b_bih, l1b_bhh, nullptr, preB,
                   MSB, 4 * H_, 2 * H_, 2 * H_, 2 * H_, 4 * H_);
  lstm_dir_kernel<<<2, 256, 0, stream>>>(preF, preB, whh1f, whh1b, h1);

  // ---- GAT: g = fc1(x), x rows m = b*S + s (plain) ----
  gemm<0, 0, true>(stream, x, fc1_W, fc1_b, nullptr, nullptr, gA,
                   MSB, G_, D_, D_, D_, G_);

  float* cur = gA; float* t1 = gB; float* t2 = gC;
  const long long sGG = (long long)S_ * G_;
  const long long sSS = (long long)S_ * S_;
  for (int l = 0; l < 4; ++l) {
    const float* Wq = wattn_W + (size_t)l * G_ * G_;
    // q = cur @ Wq^T + bq
    gemm<0, 0, true>(stream, cur, Wq, wattn_b + l * G_, nullptr, nullptr, t2,
                     MSB, G_, G_, G_, G_, G_);
    // scores[b] = q[b] @ cur[b]^T  (batched)
    gemm<0, 0, true>(stream, t2, cur, nullptr, nullptr, nullptr, scores,
                     S_, S_, G_, G_, G_, S_, sGG, sGG, sSS, B_);
    // sigmoid + eye + adj mask + row-normalize
    attn_mask_norm_kernel<<<dim3(S_, B_), 256, 0, stream>>>(scores, adj);
    // gT[b] = cur[b]^T so aggregation B-fragments are contiguous
    transpose_bsg_kernel<<<dim3(S_ / 32, G_ / 32, B_), dim3(32, 8), 0, stream>>>(cur, gT);
    // aggregate: t1[b] = a[b] @ cur[b]  ==  a[b] @ (gT[b])^T
    gemm<0, 0, true>(stream, scores, gT, nullptr, nullptr, nullptr, t1,
                     S_, G_, S_, S_, S_, G_, sSS, (long long)G_ * S_, sGG, B_);
    // t2 = relu(lin0(t1))
    gemm<0, 1, true>(stream, t1, lin0_W + (size_t)l * G_ * G_, lin0_b + l * G_,
                     nullptr, nullptr, t2, MSB, G_, G_, G_, G_, G_);
    // t1 = relu(lin1(t2)) + cur   (residual)
    gemm<0, 1, true>(stream, t2, lin1_W + (size_t)l * G_ * G_, lin1_b + l * G_,
                     nullptr, cur, t1, MSB, G_, G_, G_, G_, G_);
    float* old = cur; cur = t1; t1 = old;
  }
  // gfin = linf(cur)
  gemm<0, 0, true>(stream, cur, linf_W, linf_b, nullptr, nullptr, t2,
                   MSB, G_, G_, G_, G_, G_);
  float* gfin = t2;

  // ---- head:  d_out = [ y (16x2) | hidden (16x512) ] ----
  float* y_out  = (float*)d_out;
  float* hidden = (float*)d_out + B_ * 2;
  head_mean_kernel<<<B_, 256, 0, stream>>>(h1, gfin, hidden);

  gemm<0, 2, true >(stream, hidden, fcA_W, fcA_b, nullptr, nullptr, fcAo,
                    B_, 128, 512, 512, 512, 128);
  gemm<0, 2, false>(stream, fcAo, fcB_W, fcB_b, nullptr, nullptr, fcBo,
                    B_, 32, 128, 128, 128, 32);
  gemm<0, 0, false>(stream, fcBo, fcC_W, fcC_b, nullptr, nullptr, y_out,
                    B_, 2, 32, 32, 32, 2);
}
